// CausalTransformer_58892591563362
// MI455X (gfx1250) — matrix operand order repaired and checked by reading, not compiled, not run on hardware
//
#include <hip/hip_runtime.h>
#include <math.h>

// ---------------- problem constants ----------------
#define B_   64
#define V_   256
#define D_   512
#define DFF_ 2048
#define L_   8
#define H_   8
#define T_   (B_ * V_)          // 16384 tokens
#define EPS_ 1e-5f

typedef __bf16 bf16;
typedef __attribute__((ext_vector_type(16))) __bf16 v16bf;
typedef __attribute__((ext_vector_type(8)))  float  v8f;
typedef unsigned int u32x4 __attribute__((ext_vector_type(4)));
typedef int v4i __attribute__((vector_size(16)));   // matches builtin's pointee type

#define AS1 __attribute__((address_space(1)))
#define AS3 __attribute__((address_space(3)))

// CDNA5 async global->LDS copy path (ASYNCcnt), guarded so both toolchains compile.
#if defined(__gfx1250__) && __has_builtin(__builtin_amdgcn_global_load_async_to_lds_b128) && \
    __has_builtin(__builtin_amdgcn_s_wait_asynccnt)
#define USE_ASYNC 1
#else
#define USE_ASYNC 0
#endif

// copy 16B global -> LDS (per-lane). Async variant uses GLOBAL_LOAD_ASYNC_TO_LDS_B128.
__device__ __forceinline__ void cp16(const bf16* g, bf16* l) {
#if USE_ASYNC
  __builtin_amdgcn_global_load_async_to_lds_b128(
      (AS1 v4i*)const_cast<bf16*>(g), (AS3 v4i*)l, 0, 0);
#else
  *(u32x4*)l = *(const u32x4*)g;
#endif
}
__device__ __forceinline__ void cp_wait() {
#if USE_ASYNC
  __builtin_amdgcn_s_wait_asynccnt(0);   // s_wait_asynccnt 0
#endif
}

union FragU { v16bf v; u32x4 q[2]; };

// ---------------- WMMA fragment loaders (CDNA5 wave32 layouts) ----------------
// A fragment 16x32 bf16 (MxK): lane L: M=L&15; K runs {0..7,16..23} (lanes<16)
// or {8..15,24..31} (lanes>=16) => two contiguous 16B loads.
__device__ __forceinline__ v16bf load_afrag(const bf16* __restrict__ base, int ld,
                                            int row, int k0, int lane) {
  int m  = lane & 15;
  int ko = (lane & 16) ? 8 : 0;
  const bf16* p = base + (size_t)(row + m) * (size_t)ld + (size_t)(k0 + ko);
  FragU f;
  f.q[0] = *(const u32x4*)(p);
  f.q[1] = *(const u32x4*)(p + 16);
  return f.v;
}
// B fragment 32x16 (KxN), B[k,n]=W[n,k], W row-major [N,K]: lane L: N=L&15,
// K runs {0..7,16..23} (lanes<16) or {8..15,24..31} (lanes>=16) => two 16B loads.
__device__ __forceinline__ v16bf load_bfrag(const bf16* __restrict__ Wb, int ldk,
                                            int n, int k0, int lane) {
  int nn = lane & 15;
  int ko = (lane & 16) ? 8 : 0;
  const bf16* p = Wb + (size_t)(n + nn) * (size_t)ldk + (size_t)(k0 + ko);
  FragU f;
  f.q[0] = *(const u32x4*)(p);
  f.q[1] = *(const u32x4*)(p + 16);
  return f.v;
}

__device__ __forceinline__ v8f wmma_bf16(v16bf a, v16bf b, v8f c) {
  return __builtin_amdgcn_wmma_f32_16x16x32_bf16(
      false, a, false, b, (short)0, c, false, false);
}

#define V8F_ZERO {0.f,0.f,0.f,0.f,0.f,0.f,0.f,0.f}

// ---------------- GEMM: C[M,N] = A[M,K] @ W[N,K]^T + bias ----------------
// Block: 256 threads = 8 waves (2 in M x 4 in N), tile 64(M) x 256(N).
// K loop staged through double-buffered LDS (A 64x32, B 256x32) via async copies.
// mode 0: fp32 C; mode 1: relu->bf16; mode 2: QKV split (q scaled 1/8, v transposed).
__global__ __launch_bounds__(256)
void gemm_kernel(const bf16* __restrict__ A, const bf16* __restrict__ W,
                 const float* __restrict__ bias,
                 float* __restrict__ Cf, bf16* __restrict__ Cb,
                 bf16* __restrict__ qb, bf16* __restrict__ kb, bf16* __restrict__ vt,
                 int M, int N, int K, int mode) {
  __shared__ __align__(16) bf16 aT[2][64 * 32];    //  8 KB
  __shared__ __align__(16) bf16 bT[2][256 * 32];   // 32 KB
  const int tid  = threadIdx.x;
  const int lane = tid & 31;
  const int wave = tid >> 5;
  const int wm = wave >> 2, wn = wave & 3;         // 2 x 4 wave grid
  const int m0 = blockIdx.x * 64;
  const int n0 = blockIdx.y * 256;
  const int wmr = wm * 32;                         // wave row offset (0/32)
  const int wnc = wn * 64;                         // wave col offset (0/64/128/192)

  v8f acc[2][4];
#pragma unroll
  for (int mi = 0; mi < 2; ++mi)
#pragma unroll
    for (int ni = 0; ni < 4; ++ni) acc[mi][ni] = (v8f)V8F_ZERO;

  auto stage = [&](int k0, int s) {
    // A tile: 64 rows x 64B -> 256 x 16B segments (1 per thread)
    {
      const int row = tid >> 2, seg = tid & 3;
      cp16(A + (size_t)(m0 + row) * K + k0 + seg * 8, &aT[s][row * 32 + seg * 8]);
    }
    // B tile: 256 rows x 64B -> 1024 x 16B segments (4 per thread)
#pragma unroll
    for (int r = 0; r < 4; ++r) {
      const int sid = tid + r * 256;
      const int row = sid >> 2, seg = sid & 3;
      cp16(W + (size_t)(n0 + row) * K + k0 + seg * 8, &bT[s][row * 32 + seg * 8]);
    }
  };

  const int nsteps = K >> 5;
  stage(0, 0);
  for (int i = 0; i < nsteps; ++i) {
    cp_wait();             // own async copies for buffer i&1 complete
    __syncthreads();       // everyone's copies visible; prior compute on other buffer done
    if (i + 1 < nsteps) stage((i + 1) * 32, (i + 1) & 1);

    const bf16* aB = &aT[i & 1][0];
    const bf16* bB = &bT[i & 1][0];
    v16bf a0 = load_afrag(aB, 32, wmr,      0, lane);
    v16bf a1 = load_afrag(aB, 32, wmr + 16, 0, lane);
#pragma unroll
    for (int ni = 0; ni < 4; ++ni) {
      v16bf bb = load_bfrag(bB, 32, wnc + ni * 16, 0, lane);
      acc[0][ni] = wmma_bf16(a0, bb, acc[0][ni]);
      acc[1][ni] = wmma_bf16(a1, bb, acc[1][ni]);
    }
  }

  // D layout: lane holds column n = tile + (lane&15); rows r + 8*(lane>>4)
  const int colL = lane & 15;
  const int rowH = (lane >> 4) * 8;
#pragma unroll
  for (int mi = 0; mi < 2; ++mi) {
#pragma unroll
    for (int ni = 0; ni < 4; ++ni) {
      const int n = n0 + wnc + ni * 16 + colL;
      const float bv = bias[n];
#pragma unroll
      for (int r = 0; r < 8; ++r) {
        const int m = m0 + wmr + mi * 16 + rowH + r;
        const float v = acc[mi][ni][r] + bv;
        if (mode == 0) {
          Cf[(size_t)m * N + n] = v;
        } else if (mode == 1) {
          Cb[(size_t)m * N + n] = (bf16)(v > 0.f ? v : 0.f);
        } else { // QKV split
          if (n < 512) {
            qb[(size_t)m * D_ + n] = (bf16)(v * 0.125f);      // fold 1/sqrt(64)
          } else if (n < 1024) {
            kb[(size_t)m * D_ + (n - 512)] = (bf16)v;
          } else {
            const int d  = n - 1024;
            const int bb = m >> 8, s = m & 255;               // S = 256
            const int hh = d >> 6, dd = d & 63;               // hd = 64
            vt[(((size_t)bb * H_ + hh) * 64 + dd) * 256 + s] = (bf16)v;
          }
        }
      }
    }
  }
}

// ---------------- attention: scores (WMMA) -> softmax (LDS) -> PV (WMMA) ----------------
// grid = B*H*2, 128 threads (4 waves), wave owns 32 query rows.
// dynamic LDS: 4 * 32 * 256 fp32 = 128 KB (CDNA5: 320 KB/WGP).
__global__ __launch_bounds__(128)
void attn_kernel(const bf16* __restrict__ qb, const bf16* __restrict__ kb,
                 const bf16* __restrict__ vt, bf16* __restrict__ ctx,
                 float* __restrict__ attn_acc) {
  extern __shared__ float smem[];
  const int lane = threadIdx.x & 31;
  const int wave = threadIdx.x >> 5;
  const int bh   = blockIdx.x >> 1;
  const int half = blockIdx.x & 1;
  const int b = bh >> 3, h = bh & 7;
  const int q0 = half * 128 + wave * 32;

  float* sc = smem + (size_t)wave * (32 * 256);

  const bf16* qbase = qb + ((size_t)(b * 256 + q0)) * D_ + h * 64;
  const bf16* kbase = kb + ((size_t)(b * 256)) * D_ + h * 64;

  const int colL = lane & 15;
  const int rowH = (lane >> 4) * 8;

  // phase 1: scores = (q/8) @ k^T
  for (int nt = 0; nt < 16; ++nt) {
    v8f acc0 = V8F_ZERO, acc1 = V8F_ZERO;
#pragma unroll
    for (int ks = 0; ks < 2; ++ks) {
      const int k0 = ks * 32;
      v16bf a0 = load_afrag(qbase, D_, 0,  k0, lane);
      v16bf a1 = load_afrag(qbase, D_, 16, k0, lane);
      v16bf bb = load_bfrag(kbase, D_, nt * 16, k0, lane);
      acc0 = wmma_bf16(a0, bb, acc0);
      acc1 = wmma_bf16(a1, bb, acc1);
    }
#pragma unroll
    for (int r = 0; r < 8; ++r) {
      sc[(rowH + r) * 256 + nt * 16 + colL]      = acc0[r];
      sc[(16 + rowH + r) * 256 + nt * 16 + colL] = acc1[r];
    }
  }
  __syncthreads();

  // phase 2: exact softmax; lane owns one row; accumulate avg_attn (mean over L*H = 64)
  {
    float* row = sc + (size_t)lane * 256;
    float mx = -1e30f;
    for (int j = 0; j < 256; ++j) mx = fmaxf(mx, row[j]);
    float sum = 0.f;
    for (int j = 0; j < 256; ++j) sum += __expf(row[j] - mx);
    const float inv = 1.f / sum;
    bf16* prow = (bf16*)row;      // in-place bf16 probs (write idx j/2 <= read idx j)
    float* outp = attn_acc + ((size_t)b * 256 + (q0 + lane)) * 256;
    for (int j = 0; j < 256; ++j) {
      const float p = __expf(row[j] - mx) * inv;
      prow[j] = (bf16)p;
      atomicAdd(outp + j, p * (1.0f / 64.0f));
    }
  }
  __syncthreads();

  // phase 3: ctx = P @ V  (P in LDS, row stride 512 halfwords; V transposed in vt)
  const bf16* pbase = (const bf16*)sc;
  const bf16* vbase = vt + ((size_t)(b * H_ + h)) * 64 * 256;

  v8f acc[2][4];
#pragma unroll
  for (int mi = 0; mi < 2; ++mi)
#pragma unroll
    for (int ni = 0; ni < 4; ++ni) acc[mi][ni] = (v8f)V8F_ZERO;

  for (int ks = 0; ks < 8; ++ks) {
    const int k0 = ks * 32;
    v16bf a0 = load_afrag(pbase, 512, 0,  k0, lane);
    v16bf a1 = load_afrag(pbase, 512, 16, k0, lane);
#pragma unroll
    for (int ni = 0; ni < 4; ++ni) {
      v16bf bb = load_bfrag(vbase, 256, ni * 16, k0, lane);
      acc[0][ni] = wmma_bf16(a0, bb, acc[0][ni]);
      acc[1][ni] = wmma_bf16(a1, bb, acc[1][ni]);
    }
  }

#pragma unroll
  for (int mi = 0; mi < 2; ++mi)
#pragma unroll
    for (int ni = 0; ni < 4; ++ni)
#pragma unroll
      for (int r = 0; r < 8; ++r) {
        const int qi = q0 + mi * 16 + rowH + r;
        const int d  = ni * 16 + colL;
        ctx[((size_t)(b * 256 + qi)) * D_ + h * 64 + d] = (bf16)acc[mi][ni][r];
      }
}

// ---------------- residual + LayerNorm (one block per token) ----------------
__global__ __launch_bounds__(256)
void resid_ln_kernel(float* __restrict__ h, bf16* __restrict__ hb,
                     const float* __restrict__ Cf,
                     const float* __restrict__ g, const float* __restrict__ bta) {
  __shared__ float red[256];
  const int t = blockIdx.x, tid = threadIdx.x;
  const size_t base = (size_t)t * D_;
  float v0 = h[base + tid]       + Cf[base + tid];
  float v1 = h[base + tid + 256] + Cf[base + tid + 256];

  red[tid] = v0 + v1; __syncthreads();
  for (int s = 128; s > 0; s >>= 1) { if (tid < s) red[tid] += red[tid + s]; __syncthreads(); }
  const float mean = red[0] * (1.f / D_);
  __syncthreads();

  const float d0 = v0 - mean, d1 = v1 - mean;
  red[tid] = d0 * d0 + d1 * d1; __syncthreads();
  for (int s = 128; s > 0; s >>= 1) { if (tid < s) red[tid] += red[tid + s]; __syncthreads(); }
  const float inv = rsqrtf(red[0] * (1.f / D_) + EPS_);

  const float y0 = d0 * inv * g[tid]       + bta[tid];
  const float y1 = d1 * inv * g[tid + 256] + bta[tid + 256];
  h[base + tid] = y0;        h[base + tid + 256] = y1;
  hb[base + tid] = (bf16)y0; hb[base + tid + 256] = (bf16)y1;
}

// ---------------- embedding: fe + ve (scalar MLP + LN) + var_emb + mask_emb ----------------
struct EmbedParams {
  const float *x, *mask, *value;
  const float *fw1, *fb1, *fw2, *fb2, *fg, *fbn;
  const float *vw1, *vb1, *vw2, *vb2, *vg, *vbn;
  const float *var_emb, *me_w, *me_b;
  float* h; bf16* hb;
};

__global__ __launch_bounds__(256)
void embed_kernel(EmbedParams P) {
  __shared__ float h1[2 * D_];
  __shared__ float red[256];
  const int t = blockIdx.x, tid = threadIdx.x;
  const int vv = t & (V_ - 1);
  float acc0 = 0.f, acc1 = 0.f;

  for (int e = 0; e < 2; ++e) {
    const float* w1 = e ? P.vw1 : P.fw1;  const float* b1 = e ? P.vb1 : P.fb1;
    const float* w2 = e ? P.vw2 : P.fw2;  const float* b2 = e ? P.vb2 : P.fb2;
    const float* g  = e ? P.vg  : P.fg;   const float* bn = e ? P.vbn : P.fbn;
    const float s = (e ? P.value : P.x)[t];

#pragma unroll
    for (int jj = 0; jj < 4; ++jj) {
      const int j = tid + jj * 256;
      const float z = s * w1[j] + b1[j];
      h1[j] = 0.5f * z * (1.f + erff(z * 0.7071067811865475f));   // exact GELU
    }
    __syncthreads();

    float dot0 = b2[tid], dot1 = b2[tid + 256];
    for (int j = 0; j < 2 * D_; ++j) {
      const float hv = h1[j];
      dot0 += hv * w2[(size_t)tid * (2 * D_) + j];
      dot1 += hv * w2[(size_t)(tid + 256) * (2 * D_) + j];
    }

    red[tid] = dot0 + dot1; __syncthreads();
    for (int sf = 128; sf > 0; sf >>= 1) { if (tid < sf) red[tid] += red[tid + sf]; __syncthreads(); }
    const float mean = red[0] * (1.f / D_);
    __syncthreads();
    const float d0 = dot0 - mean, d1 = dot1 - mean;
    red[tid] = d0 * d0 + d1 * d1; __syncthreads();
    for (int sf = 128; sf > 0; sf >>= 1) { if (tid < sf) red[tid] += red[tid + sf]; __syncthreads(); }
    const float inv = rsqrtf(red[0] * (1.f / D_) + EPS_);
    acc0 += d0 * inv * g[tid]       + bn[tid];
    acc1 += d1 * inv * g[tid + 256] + bn[tid + 256];
    __syncthreads();
  }

  const float mval = P.mask[t];
  acc0 += P.var_emb[(size_t)vv * D_ + tid]       + mval * P.me_w[tid]       + P.me_b[tid];
  acc1 += P.var_emb[(size_t)vv * D_ + tid + 256] + mval * P.me_w[tid + 256] + P.me_b[tid + 256];

  const size_t base = (size_t)t * D_;
  P.h[base + tid] = acc0;        P.h[base + tid + 256] = acc1;
  P.hb[base + tid] = (bf16)acc0; P.hb[base + tid + 256] = (bf16)acc1;
}

// ---------------- output head ----------------
__global__ __launch_bounds__(256)
void pred_kernel(const float* __restrict__ h, const float* __restrict__ ow,
                 const float* __restrict__ ob, float* __restrict__ pred) {
  const int wave = threadIdx.x >> 5, lane = threadIdx.x & 31;
  const int t = blockIdx.x * 8 + wave;
  float s = 0.f;
#pragma unroll
  for (int i = 0; i < 16; ++i) s += h[(size_t)t * D_ + lane + i * 32] * ow[lane + i * 32];
#pragma unroll
  for (int m = 16; m; m >>= 1) s += __shfl_xor(s, m, 32);
  if (lane == 0) pred[t] = s + ob[0];
}

// ---------------- utilities ----------------
__global__ void cvt_bf16_kernel(const float* __restrict__ in, bf16* __restrict__ out, size_t n) {
  for (size_t i = (size_t)blockIdx.x * blockDim.x + threadIdx.x; i < n;
       i += (size_t)gridDim.x * blockDim.x)
    out[i] = (bf16)in[i];
}
__global__ void zero_f32_kernel(float* __restrict__ p, size_t n) {
  for (size_t i = (size_t)blockIdx.x * blockDim.x + threadIdx.x; i < n;
       i += (size_t)gridDim.x * blockDim.x)
    p[i] = 0.f;
}

// ---------------- host launcher ----------------
extern "C" void kernel_launch(void* const* d_in, const int* in_sizes, int n_in,
                              void* d_out, int out_size, void* d_ws, size_t ws_size,
                              hipStream_t stream) {
  const float* x      = (const float*)d_in[0];
  const float* mask   = (const float*)d_in[1];
  const float* value  = (const float*)d_in[2];
  const float* fe_w1  = (const float*)d_in[3];
  const float* fe_b1  = (const float*)d_in[4];
  const float* fe_w2  = (const float*)d_in[5];
  const float* fe_b2  = (const float*)d_in[6];
  const float* fe_g   = (const float*)d_in[7];
  const float* fe_bn  = (const float*)d_in[8];
  const float* ve_w1  = (const float*)d_in[9];
  const float* ve_b1  = (const float*)d_in[10];
  const float* ve_w2  = (const float*)d_in[11];
  const float* ve_b2  = (const float*)d_in[12];
  const float* ve_g   = (const float*)d_in[13];
  const float* ve_bn  = (const float*)d_in[14];
  const float* var_emb= (const float*)d_in[15];
  const float* me_w   = (const float*)d_in[16];
  const float* me_b   = (const float*)d_in[17];
  const float* Wqkv   = (const float*)d_in[18];
  const float* bqkv   = (const float*)d_in[19];
  const float* Wo     = (const float*)d_in[20];
  const float* bo     = (const float*)d_in[21];
  const float* W1     = (const float*)d_in[22];
  const float* b1     = (const float*)d_in[23];
  const float* W2     = (const float*)d_in[24];
  const float* b2     = (const float*)d_in[25];
  const float* ln1_g  = (const float*)d_in[26];
  const float* ln1_b  = (const float*)d_in[27];
  const float* ln2_g  = (const float*)d_in[28];
  const float* ln2_b  = (const float*)d_in[29];
  const float* out_w  = (const float*)d_in[30];
  const float* out_b  = (const float*)d_in[31];
  (void)in_sizes; (void)n_in; (void)out_size; (void)ws_size;

  float* pred     = (float*)d_out;            // [B,V]
  float* avg_attn = (float*)d_out + T_;       // [B,V,V]

  char* ws = (char*)d_ws;
  auto carve = [&](size_t bytes) { char* p = ws; ws += (bytes + 255) & ~(size_t)255; return p; };
  float* h   = (float*)carve((size_t)T_ * D_ * 4);
  bf16*  hb  = (bf16*) carve((size_t)T_ * D_ * 2);
  bf16*  qb  = (bf16*) carve((size_t)T_ * D_ * 2);
  bf16*  kb  = (bf16*) carve((size_t)T_ * D_ * 2);
  bf16*  vt  = (bf16*) carve((size_t)T_ * D_ * 2);
  bf16*  ctx = (bf16*) carve((size_t)T_ * D_ * 2);
  float* Cf  = (float*)carve((size_t)T_ * D_ * 4);
  bf16*  f1  = (bf16*) carve((size_t)T_ * DFF_ * 2);
  bf16*  wqkvB = (bf16*)carve((size_t)L_ * 3 * D_ * D_ * 2);
  bf16*  woB   = (bf16*)carve((size_t)L_ * D_ * D_ * 2);
  bf16*  w1B   = (bf16*)carve((size_t)L_ * DFF_ * D_ * 2);
  bf16*  w2B   = (bf16*)carve((size_t)L_ * D_ * DFF_ * 2);

  (void)hipFuncSetAttribute((const void*)attn_kernel,
                            hipFuncAttributeMaxDynamicSharedMemorySize, 131072);

  cvt_bf16_kernel<<<4096, 256, 0, stream>>>(Wqkv, wqkvB, (size_t)L_ * 3 * D_ * D_);
  cvt_bf16_kernel<<<4096, 256, 0, stream>>>(Wo,   woB,   (size_t)L_ * D_ * D_);
  cvt_bf16_kernel<<<4096, 256, 0, stream>>>(W1,   w1B,   (size_t)L_ * DFF_ * D_);
  cvt_bf16_kernel<<<4096, 256, 0, stream>>>(W2,   w2B,   (size_t)L_ * D_ * DFF_);

  zero_f32_kernel<<<4096, 256, 0, stream>>>(avg_attn, (size_t)B_ * V_ * V_);

  EmbedParams ep{ x, mask, value,
                  fe_w1, fe_b1, fe_w2, fe_b2, fe_g, fe_bn,
                  ve_w1, ve_b1, ve_w2, ve_b2, ve_g, ve_bn,
                  var_emb, me_w, me_b, h, hb };
  embed_kernel<<<T_, 256, 0, stream>>>(ep);

  for (int l = 0; l < L_; ++l) {
    // QKV projection (N = 1536), split epilogue
    gemm_kernel<<<dim3(T_ / 64, 6), 256, 0, stream>>>(
        hb, wqkvB + (size_t)l * 3 * D_ * D_, bqkv + (size_t)l * 3 * D_,
        nullptr, nullptr, qb, kb, vt, T_, 3 * D_, D_, 2);

    attn_kernel<<<B_ * H_ * 2, 128, 131072, stream>>>(qb, kb, vt, ctx, avg_attn);

    // Wo projection (fp32 out)
    gemm_kernel<<<dim3(T_ / 64, 2), 256, 0, stream>>>(
        ctx, woB + (size_t)l * D_ * D_, bo + (size_t)l * D_,
        Cf, nullptr, nullptr, nullptr, nullptr, T_, D_, D_, 0);

    resid_ln_kernel<<<T_, 256, 0, stream>>>(h, hb, Cf,
        ln1_g + (size_t)l * D_, ln1_b + (size_t)l * D_);

    // FFN1 (relu -> bf16)
    gemm_kernel<<<dim3(T_ / 64, 8), 256, 0, stream>>>(
        hb, w1B + (size_t)l * DFF_ * D_, b1 + (size_t)l * DFF_,
        nullptr, f1, nullptr, nullptr, nullptr, T_, DFF_, D_, 1);

    // FFN2 (fp32 out)
    gemm_kernel<<<dim3(T_ / 64, 2), 256, 0, stream>>>(
        f1, w2B + (size_t)l * D_ * DFF_, b2 + (size_t)l * D_,
        Cf, nullptr, nullptr, nullptr, nullptr, T_, D_, DFF_, 0);

    resid_ln_kernel<<<T_, 256, 0, stream>>>(h, hb, Cf,
        ln2_g + (size_t)l * D_, ln2_b + (size_t)l * D_);
  }

  pred_kernel<<<T_ / 8, 256, 0, stream>>>(h, out_w, out_b, pred);
}
